// SlidingWindowGQA_31937376813255
// MI455X (gfx1250) — matrix operand order repaired and checked by reading, not compiled, not run on hardware
//
#include <hip/hip_runtime.h>

// Sliding-window GQA block for gfx1250 (MI455X), wave32 + WMMA bf16 + TDM.
// Pipeline: cvt/transpose -> QKV GEMM (TDM double-buffered LDS, WMMA) ->
// RoPE/layout -> windowed attention (WMMA QK^T, LDS softmax, WMMA PV) ->
// output GEMM (WMMA).  Workspace requirement: ~100 MB of d_ws.

#define B_    2
#define S_    2048
#define DM_   2048
#define H_    16
#define KVH_  4
#define HD_   128
#define WIN_  512
#define NQKV_ 3072
#define KD_   2048
#define QT_   64
#define KSPAN_ 640   // 512 window + 64 tile skew, rounded to 20 chunks of 32

typedef __attribute__((ext_vector_type(16))) __bf16 v16bf;
typedef __attribute__((ext_vector_type(8)))  __bf16 v8bf;
typedef __attribute__((ext_vector_type(8)))  float  v8f;
typedef __attribute__((ext_vector_type(4)))  unsigned v4u;
typedef __attribute__((ext_vector_type(8)))  int      v8i;
typedef __attribute__((ext_vector_type(4)))  int      v4i;

static __device__ __forceinline__ v16bf cat8(v8bf lo, v8bf hi) {
  return __builtin_shufflevector(lo, hi, 0,1,2,3,4,5,6,7,8,9,10,11,12,13,14,15);
}
static __device__ __forceinline__ v8bf ld8(const __bf16* p) { return *(const v8bf*)p; }

static __device__ __forceinline__ v8f wmma_bf16(v16bf a, v16bf b, v8f c) {
  return __builtin_amdgcn_wmma_f32_16x16x32_bf16(false, a, false, b, (short)0, c, false, false);
}

// ---------------- stage 0: precision conversion / weight transpose ----------
__global__ __launch_bounds__(256)
void k_cvt(const float* __restrict__ s, __bf16* __restrict__ d, int n) {
  int i = blockIdx.x * 256 + threadIdx.x;
  if (i < n) d[i] = (__bf16)s[i];
}

// W [K][N] f32 row-major  ->  Wt [N][K] bf16 (so WMMA B-frags are contiguous)
__global__ __launch_bounds__(256)
void k_tconv(const float* __restrict__ w, __bf16* __restrict__ wt, int K, int N) {
  int i = blockIdx.x * 256 + threadIdx.x;
  if (i < K * N) {
    int k = i / N, n = i - k * N;
    wt[(size_t)n * K + k] = (__bf16)w[i];
  }
}

// ---------------- TDM: 128-row x 32-col bf16 tile -> LDS --------------------
// D# per CDNA5 ISA. Pads +4 DWORDs after every 16 DWORDs
// stored, turning the packed 64B tile row into an 80B LDS pitch (LSTR=40).
// 6-arg builtin form (amdgpu-toolchain / clang-23).
#define LSTR 40
static __device__ __forceinline__ void tdm_tile_128x32(unsigned lds_byte,
                                                       const __bf16* g,
                                                       unsigned stride_elems) {
  unsigned long long ga = (unsigned long long)(size_t)g;
  v4u g0;
  g0[0] = 1u;                                        // count=1 (valid user D#)
  g0[1] = lds_byte;                                  // lds_addr
  g0[2] = (unsigned)ga;                              // global_addr[31:0]
  g0[3] = (unsigned)((ga >> 32) & 0x01ffffffu) | (2u << 30);  // addr[56:32]|type=2
  v8i g1;
  g1[0] = (int)((1u << 16)        // data_size = 2 bytes
              | (1u << 20)        // pad_enable
              | (3u << 22)        // pad_interval: every 16 DWORDs (64B row)
              | (3u << 25));      // pad_amount: 4 DWORDs (16B) -> 80B pitch
  g1[1] = (int)((32u & 0xffffu) << 16);              // tensor_dim0[15:0] = 32
  g1[2] = (int)((128u & 0xffffu) << 16);             // dim0 hi=0 | tensor_dim1=128
  g1[3] = (int)((32u & 0xffffu) << 16);              // dim1 hi=0 | tile_dim0=32
  g1[4] = (int)(128u);                               // tile_dim1=128, tile_dim2=0
  g1[5] = (int)stride_elems;                         // tensor_dim0_stride[31:0]
  g1[6] = 0;                                         // stride hi, dim1_stride lo
  g1[7] = 0;
  v4i z4 = {0, 0, 0, 0};
  v8i z8 = {0, 0, 0, 0, 0, 0, 0, 0};
  __builtin_amdgcn_tensor_load_to_lds(g0, g1, z4, z4, z8, 0);
}

// ---------------- stage 1/4: tiled WMMA GEMM with TDM staging ---------------
// C[M,N] = A[M,K] * Bt[N,K]^T ; workgroup tile 128x128, 8 waves (2x4) of 64x32.
// Wave 0 drives the Tensor Data Mover; tiles double-buffered in LDS (40 KB).
__global__ __launch_bounds__(256)
void k_gemm(const __bf16* __restrict__ A, const __bf16* __restrict__ Bt,
            __bf16* __restrict__ Cb, float* __restrict__ Cf,
            int M, int N, int K)
{
  __shared__ __bf16 SH[4 * 128 * LSTR];   // [buf][A|B][128*40] = 40 KB
  const int m0 = blockIdx.y * 128, n0 = blockIdx.x * 128;
  const int tid = threadIdx.x, lane = tid & 31, wave = tid >> 5;
  const int wm = wave >> 2, wn = wave & 3;
  const int half = lane >> 4, lr = lane & 15;
  const int NK = K >> 5;                  // K-tiles of 32

  // The TDM builtin carries no pointer operand, so the compiler believes SH is
  // never written and folds the ds_load fragment reads to undef (round-3 asm).
  // Defeat that with a never-executed store at an opaque offset: zoff is 0 at
  // runtime, but the compiler must assume SH[zoff] may be written, and the
  // unknown offset may-aliases every fragment load. Accesses stay on SH itself
  // so address-space inference still emits native ds_load_b128 (round-4 fix:
  // laundering the pointer degraded the loads to flat_load_b128).
  unsigned zoff = 0;
  asm volatile("" : "+s"(zoff));
  if (zoff) SH[zoff & 16383] = (__bf16)0.0f;

  v8f acc[4][2] = {};

  // prologue: TDM fetch of tile 0 into buffer 0
  if (wave == 0) {
    tdm_tile_128x32(0u,      A  + (size_t)m0 * K, (unsigned)K);
    tdm_tile_128x32(10240u,  Bt + (size_t)n0 * K, (unsigned)K);
  }

  for (int kt = 0; kt < NK; ++kt) {
    const int cur = kt & 1;
    if (wave == 0) {
      if (kt + 1 < NK) {                  // kick off next tile into other buffer
        const int nxt = cur ^ 1;
        const int kn = (kt + 1) << 5;
        tdm_tile_128x32((unsigned)(nxt * 20480),         A  + (size_t)m0 * K + kn, (unsigned)K);
        tdm_tile_128x32((unsigned)(nxt * 20480 + 10240), Bt + (size_t)n0 * K + kn, (unsigned)K);
        __builtin_amdgcn_s_wait_tensorcnt(2);   // current tile landed in LDS
      } else {
        __builtin_amdgcn_s_wait_tensorcnt(0);
      }
      asm volatile("" ::: "memory");      // TDM's LDS writes are now visible
    }
    __syncthreads();                      // release workgroup onto fresh tile

    const __bf16* Ash = SH + cur * 10240;
    const __bf16* Bsh = Ash + 5120;
    v16bf af[4];
    for (int mf = 0; mf < 4; ++mf) {
      const __bf16* p = Ash + (wm * 64 + mf * 16 + lr) * LSTR;
      af[mf] = cat8(ld8(p + half * 8), ld8(p + 16 + half * 8));   // A-frag layout
    }
    for (int nf = 0; nf < 2; ++nf) {
      const __bf16* p = Bsh + (wn * 32 + nf * 16 + lr) * LSTR;
      v16bf bfr = cat8(ld8(p + half * 8), ld8(p + 16 + half * 8)); // B-frag layout
      for (int mf = 0; mf < 4; ++mf)
        acc[mf][nf] = wmma_bf16(af[mf], bfr, acc[mf][nf]);
    }
    __syncthreads();                      // done reading before TDM reuses buffer
  }

  for (int mf = 0; mf < 4; ++mf)
    for (int nf = 0; nf < 2; ++nf)
      for (int e = 0; e < 8; ++e) {
        int r = m0 + wm * 64 + mf * 16 + e + half * 8;
        int c = n0 + wn * 32 + nf * 16 + lr;
        if (Cb) Cb[(size_t)r * N + c] = (__bf16)acc[mf][nf][e];
        else    Cf[(size_t)r * N + c] = acc[mf][nf][e];
      }
}

// ---------------- stage 2: RoPE + layout transform --------------------------
__global__ __launch_bounds__(256)
void k_rope(const __bf16* __restrict__ QKV, __bf16* __restrict__ Qr,
            __bf16* __restrict__ Kr, __bf16* __restrict__ Vt)
{
  const int row = blockIdx.x;              // b*S + s
  const int b = row / S_, s = row - b * S_;
  const __bf16* src = QKV + (size_t)row * NQKV_;
  for (int idx = threadIdx.x; idx < NQKV_; idx += 256) {
    if (idx < 2560) {                      // Q (0..2047) or K (2048..2559)
      int base = (idx < 2048) ? 0 : 2048;
      int t = idx - base;
      int hh = t >> 7, d = t & 127;
      float v  = (float)src[idx];
      int  dp  = (d < 64) ? d + 64 : d - 64;
      float pv = (float)src[base + hh * 128 + dp];
      float inv = __powf(10000.0f, -(float)(d & 63) * (1.0f / 64.0f));
      float ang = (float)s * inv;
      float sn, cs;
      __sincosf(ang, &sn, &cs);
      float o = (d < 64) ? (v * cs - pv * sn) : (v * cs + pv * sn);
      if (idx < 2048)
        Qr[((size_t)(b * H_ + hh) * S_ + s) * HD_ + d] = (__bf16)o;
      else
        Kr[((size_t)(b * KVH_ + hh) * S_ + s) * HD_ + d] = (__bf16)o;
    } else {                               // V: transpose to [kvh][d][s]
      int t = idx - 2560;
      int hh = t >> 7, d = t & 127;
      Vt[((size_t)(b * KVH_ + hh) * HD_ + d) * S_ + s] = src[idx];
    }
  }
}

// ---------------- stage 3: sliding-window attention -------------------------
// One workgroup per (b, h, 64-query tile). 160 KB LDS score slab (two-pass
// softmax -- no online rescaling needed since the whole window fits in LDS).
__global__ __launch_bounds__(256)
void k_attn(const __bf16* __restrict__ Qr, const __bf16* __restrict__ Kr,
            const __bf16* __restrict__ Vt, __bf16* __restrict__ AO)
{
  extern __shared__ float Sc[];            // [QT_][KSPAN_] f32, reused as bf16 P
  const int bid = blockIdx.x;
  const int qt = bid & 31, h = (bid >> 5) & 15, b = bid >> 9;
  const int kvh = h >> 2;                  // GQA: 4 query heads per KV head
  const int q0 = qt * QT_;
  const int kbase = q0 - 512;              // keys [kbase, kbase+639] cover window
  const __bf16* Q  = Qr + (size_t)(b * H_ + h) * S_ * HD_;
  const __bf16* Kp = Kr + (size_t)(b * KVH_ + kvh) * S_ * HD_;
  const __bf16* Vp = Vt + (size_t)(b * KVH_ + kvh) * HD_ * S_;   // [128][S]
  const int tid = threadIdx.x, lane = tid & 31, wave = tid >> 5;
  const int half = lane >> 4, lr = lane & 15;
  const float scale = 0.08838834764831845f;  // 1/sqrt(128)

  // Q fragments register-resident for whole pass 1: [mf][kstep], 128 VGPRs
  v16bf qf[4][4];
  for (int mf = 0; mf < 4; ++mf) {
    const __bf16* p = Q + (size_t)(q0 + mf * 16 + lr) * HD_;
    for (int ks = 0; ks < 4; ++ks)
      qf[mf][ks] = cat8(ld8(p + ks * 32 + half * 8), ld8(p + ks * 32 + 16 + half * 8));
  }

  // ---- pass 1: S = Q K^T (masked, scaled) into LDS. Waves stripe key-chunks.
  for (int c = wave; c < 20; c += 8) {
    v8f sacc[4][2] = {};
    for (int nf = 0; nf < 2; ++nf) {
      int key = kbase + c * 32 + nf * 16 + lr;
      int ksafe = key < 0 ? 0 : (key > S_ - 1 ? S_ - 1 : key);   // masked later
      const __bf16* p = Kp + (size_t)ksafe * HD_;
      for (int ks = 0; ks < 4; ++ks) {
        v16bf bfr = cat8(ld8(p + ks * 32 + half * 8), ld8(p + ks * 32 + 16 + half * 8));
        for (int mf = 0; mf < 4; ++mf)
          sacc[mf][nf] = wmma_bf16(qf[mf][ks], bfr, sacc[mf][nf]);
      }
    }
    for (int mf = 0; mf < 4; ++mf)
      for (int nf = 0; nf < 2; ++nf) {
        int kj = kbase + c * 32 + nf * 16 + lr;
        for (int e = 0; e < 8; ++e) {
          int mrow = mf * 16 + e + half * 8;
          int qi = q0 + mrow;
          bool ok = (kj >= 0) & (kj <= qi) & (kj > qi - WIN_);   // causal+window
          Sc[mrow * KSPAN_ + c * 32 + nf * 16 + lr] = ok ? sacc[mf][nf][e] * scale : -1.0e30f;
        }
      }
  }
  __syncthreads();

  // ---- pass 2: row softmax; write P back in place as bf16 (reads precede writes)
  for (int r = wave; r < QT_; r += 8) {
    float* rowp = Sc + r * KSPAN_;
    float ev[20];
    float mx = -1.0e30f;
    for (int i = 0; i < 20; ++i) { ev[i] = rowp[lane + i * 32]; mx = fmaxf(mx, ev[i]); }
    for (int off = 16; off >= 1; off >>= 1) mx = fmaxf(mx, __shfl_xor(mx, off, 32));
    float sum = 0.0f;
    for (int i = 0; i < 20; ++i) { ev[i] = __expf(ev[i] - mx); sum += ev[i]; }
    for (int off = 16; off >= 1; off >>= 1) sum += __shfl_xor(sum, off, 32);
    float rinv = 1.0f / sum;                 // diagonal always allowed -> sum > 0
    __bf16* Pr = (__bf16*)rowp;
    for (int i = 0; i < 20; ++i) Pr[lane + i * 32] = (__bf16)(ev[i] * rinv);
  }
  __syncthreads();

  // ---- pass 3: O = P V ; wave w owns head-dim columns [16w, 16w+16)
  v8f oacc[4] = {};
  const int dcol = wave * 16;
  for (int c = 0; c < 20; ++c) {
    int kj0 = kbase + c * 32 + half * 16;
    int kc = kj0 < 0 ? 0 : (kj0 > S_ - 16 ? S_ - 16 : kj0);      // P==0 off-window
    const __bf16* p = Vp + (size_t)(dcol + lr) * S_ + kc;
    v16bf bfr = cat8(ld8(p), ld8(p + 8));
    for (int mf = 0; mf < 4; ++mf) {
      const __bf16* pp = (const __bf16*)(Sc + (mf * 16 + lr) * KSPAN_);
      v16bf afr = cat8(ld8(pp + c * 32 + half * 16), ld8(pp + c * 32 + half * 16 + 8));
      oacc[mf] = wmma_bf16(afr, bfr, oacc[mf]);
    }
  }
  for (int mf = 0; mf < 4; ++mf)
    for (int e = 0; e < 8; ++e) {
      int qi = q0 + mf * 16 + e + half * 8;
      AO[(size_t)(b * S_ + qi) * DM_ + h * HD_ + dcol + lr] = (__bf16)oacc[mf][e];
    }
}

// ---------------- host-side orchestration -----------------------------------
extern "C" void kernel_launch(void* const* d_in, const int* in_sizes, int n_in,
                              void* d_out, int out_size, void* d_ws, size_t ws_size,
                              hipStream_t stream)
{
  (void)in_sizes; (void)n_in; (void)out_size; (void)ws_size;
  const float* x  = (const float*)d_in[0];
  const float* wq = (const float*)d_in[1];
  const float* wk = (const float*)d_in[2];
  const float* wv = (const float*)d_in[3];
  const float* wo = (const float*)d_in[4];
  float* out = (float*)d_out;
  char* ws = (char*)d_ws;

  const int M = B_ * S_;                                   // 4096
  __bf16* xb    = (__bf16*)(ws);                           // 16 MB  [M][2048]
  __bf16* wqkvt = (__bf16*)(ws + (size_t)16  * (1u<<20));  // 12 MB  [3072][2048]
  __bf16* wot   = (__bf16*)(ws + (size_t)28  * (1u<<20));  //  8 MB  [2048][2048]
  __bf16* qkv   = (__bf16*)(ws + (size_t)36  * (1u<<20));  // 24 MB  [M][3072]
  __bf16* qr    = (__bf16*)(ws + (size_t)60  * (1u<<20));  // 16 MB  [B][H][S][128]
  __bf16* kr    = (__bf16*)(ws + (size_t)76  * (1u<<20));  //  4 MB  [B][KVH][S][128]
  __bf16* vt    = (__bf16*)(ws + (size_t)80  * (1u<<20));  //  4 MB  [B][KVH][128][S]
  __bf16* ao    = (__bf16*)(ws + (size_t)84  * (1u<<20));  // 16 MB  [M][2048]

  // 0) precision / layout prep
  k_cvt<<<(M * KD_ + 255) / 256, 256, 0, stream>>>(x, xb, M * KD_);
  k_tconv<<<(2048 * 2048 + 255) / 256, 256, 0, stream>>>(wq, wqkvt, 2048, 2048);
  k_tconv<<<(2048 * 512  + 255) / 256, 256, 0, stream>>>(wk, wqkvt + (size_t)2048 * 2048, 2048, 512);
  k_tconv<<<(2048 * 512  + 255) / 256, 256, 0, stream>>>(wv, wqkvt + (size_t)2560 * 2048, 2048, 512);
  k_tconv<<<(2048 * 2048 + 255) / 256, 256, 0, stream>>>(wo, wot, 2048, 2048);

  // 1) fused QKV projection (bf16 WMMA, f32 accum, TDM-staged tiles)
  dim3 g1(NQKV_ / 128, M / 128);
  k_gemm<<<g1, 256, 0, stream>>>(xb, wqkvt, qkv, nullptr, M, NQKV_, KD_);

  // 2) RoPE + attention layouts
  k_rope<<<M, 256, 0, stream>>>(qkv, qr, kr, vt);

  // 3) sliding-window GQA attention (160 KB dynamic LDS score slab)
  k_attn<<<B_ * H_ * (S_ / QT_), 256, QT_ * KSPAN_ * sizeof(float), stream>>>(qr, kr, vt, ao);

  // 4) output projection -> f32 d_out
  dim3 g2(DM_ / 128, M / 128);
  k_gemm<<<g2, 256, 0, stream>>>(ao, wot, nullptr, out, M, DM_, KD_);
}
